// CodonPredictionModel_28673201668513
// MI455X (gfx1250) — compile-verified, matrix operand order checked by reading, not run on hardware
//
#include <hip/hip_runtime.h>

// ---------------------------------------------------------------------------
// Model dims
// ---------------------------------------------------------------------------
enum : int { Bn = 8, Sn = 512, Hn = 768, Ln = 12, NHn = 12, DHn = 64,
             FFn = 3072, Mn = Bn * Sn, NC = 61, VOCAB = 25 };

typedef __attribute__((ext_vector_type(16))) __bf16 v16bf;
typedef __attribute__((ext_vector_type(8)))  float  v8f;
typedef int v4i __attribute__((vector_size(16)));

// ---------------------------------------------------------------------------
// Helpers
// ---------------------------------------------------------------------------
__device__ inline unsigned short f2bf(float f) {
    union { float f; unsigned u; } v; v.f = f;
    unsigned r = v.u + 0x7FFFu + ((v.u >> 16) & 1u);   // round-to-nearest-even
    return (unsigned short)(r >> 16);
}

__device__ inline v8f v8f_zero() {
    v8f z;
#pragma unroll
    for (int i = 0; i < 8; ++i) z[i] = 0.0f;
    return z;
}

// ---- CDNA5 async global->LDS copy (ASYNCcnt-tracked), b128 granularity ----
__device__ inline void async_b128(const void* g, void* l) {
#if __has_builtin(__builtin_amdgcn_global_load_async_to_lds_b128)
    __builtin_amdgcn_global_load_async_to_lds_b128(
        (__attribute__((address_space(1))) v4i*)g,
        (__attribute__((address_space(3))) v4i*)l, 0, 0);
#else
    unsigned lo = (unsigned)(unsigned long long)l;   // LDS byte offset
    asm volatile("global_load_async_to_lds_b128 %0, %1, off"
                 :: "v"(lo), "v"(g) : "memory");
#endif
}

__device__ inline void wait_async0() {
#if __has_builtin(__builtin_amdgcn_s_wait_asynccnt)
    __builtin_amdgcn_s_wait_asynccnt(0);
#else
    asm volatile("s_wait_asynccnt 0x0" ::: "memory");
#endif
}

__device__ inline void wait_async4() {
#if __has_builtin(__builtin_amdgcn_s_wait_asynccnt)
    __builtin_amdgcn_s_wait_asynccnt(4);
#else
    asm volatile("s_wait_asynccnt 0x4" ::: "memory");
#endif
}

// Load one 16x32 bf16 fragment (A layout; B uses the same layout on the
// [n][k] buffer).  ISA 16-bit A layout:
//   lanes 0-15 : row = base+lane,    K = {kb+0..7, kb+16..23}
//   lanes16-31 : row = base+lane-16, K = {kb+8..15, kb+24..31}
__device__ inline v16bf load_frag(const unsigned short* buf, int ld,
                                  int rowBase, int kb, int lane) {
    int half = lane >> 4;
    int r    = rowBase + (lane & 15);
    int k0   = kb + (half ? 8 : 0);
    union { v16bf v; uint4 q[2]; } f;
    f.q[0] = *(const uint4*)(buf + (size_t)r * ld + k0);
    f.q[1] = *(const uint4*)(buf + (size_t)r * ld + k0 + 16);
    return f.v;
}

__device__ inline v8f wmma_bf16(v16bf a, v16bf b, v8f c) {
    return __builtin_amdgcn_wmma_f32_16x16x32_bf16(
        false, a, false, b, (short)0, c, false, false);
}

// ---------------------------------------------------------------------------
// Weight prep: WT[n][k] = bf16(W[k][n]); rows n >= N zero-filled (Npad rows)
// ---------------------------------------------------------------------------
__global__ __launch_bounds__(256)
void convert_wt_kernel(const float* __restrict__ W,
                       unsigned short* __restrict__ WT, int N, int K) {
    size_t idx = (size_t)blockIdx.x * 256 + threadIdx.x;
    int n = (int)(idx / (unsigned)K);
    int k = (int)(idx - (size_t)n * K);
    float v = (n < N) ? W[(size_t)k * N + n] : 0.f;
    WT[idx] = f2bf(v);
}

// ---------------------------------------------------------------------------
// Embedding + LayerNorm : one block (256 thr) per token
// ---------------------------------------------------------------------------
__global__ __launch_bounds__(256)
void embed_ln_kernel(const int* __restrict__ x,
                     const float* __restrict__ wemb,
                     const float* __restrict__ pemb,
                     const float* __restrict__ g,
                     const float* __restrict__ bta,
                     float* __restrict__ h,
                     unsigned short* __restrict__ hb) {
    int tok = blockIdx.x, s = tok & (Sn - 1), tid = threadIdx.x;
    int id = x[tok];
    float y[3], sum = 0.f, sq = 0.f;
#pragma unroll
    for (int i = 0; i < 3; ++i) {
        int j = tid + i * 256;
        y[i] = wemb[id * Hn + j] + pemb[s * Hn + j];
        sum += y[i]; sq += y[i] * y[i];
    }
    __shared__ float r1[8], r2[8];
    int lane = tid & 31, wave = tid >> 5;
#pragma unroll
    for (int o = 16; o; o >>= 1) { sum += __shfl_xor(sum, o, 32); sq += __shfl_xor(sq, o, 32); }
    if (!lane) { r1[wave] = sum; r2[wave] = sq; }
    __syncthreads();
    if (wave == 0) {
        float a = (lane < 8) ? r1[lane] : 0.f, c = (lane < 8) ? r2[lane] : 0.f;
#pragma unroll
        for (int o = 4; o; o >>= 1) { a += __shfl_xor(a, o, 32); c += __shfl_xor(c, o, 32); }
        if (!lane) { r1[0] = a; r2[0] = c; }
    }
    __syncthreads();
    float mean = r1[0] * (1.f / Hn);
    float var  = r2[0] * (1.f / Hn) - mean * mean;
    float inv  = rsqrtf(var + 1e-12f);
#pragma unroll
    for (int i = 0; i < 3; ++i) {
        int j = tid + i * 256;
        float v = (y[i] - mean) * inv * g[j] + bta[j];
        h [(size_t)tok * Hn + j] = v;
        hb[(size_t)tok * Hn + j] = f2bf(v);
    }
}

// ---------------------------------------------------------------------------
// Residual add + LayerNorm
// ---------------------------------------------------------------------------
__global__ __launch_bounds__(256)
void add_ln_kernel(const float* __restrict__ res,
                   const float* __restrict__ tmp,
                   const float* __restrict__ g,
                   const float* __restrict__ bta,
                   float* __restrict__ h,
                   unsigned short* __restrict__ hb) {
    int tok = blockIdx.x, tid = threadIdx.x;
    size_t base = (size_t)tok * Hn;
    float y[3], sum = 0.f, sq = 0.f;
#pragma unroll
    for (int i = 0; i < 3; ++i) {
        int j = tid + i * 256;
        y[i] = res[base + j] + tmp[base + j];
        sum += y[i]; sq += y[i] * y[i];
    }
    __shared__ float r1[8], r2[8];
    int lane = tid & 31, wave = tid >> 5;
#pragma unroll
    for (int o = 16; o; o >>= 1) { sum += __shfl_xor(sum, o, 32); sq += __shfl_xor(sq, o, 32); }
    if (!lane) { r1[wave] = sum; r2[wave] = sq; }
    __syncthreads();
    if (wave == 0) {
        float a = (lane < 8) ? r1[lane] : 0.f, c = (lane < 8) ? r2[lane] : 0.f;
#pragma unroll
        for (int o = 4; o; o >>= 1) { a += __shfl_xor(a, o, 32); c += __shfl_xor(c, o, 32); }
        if (!lane) { r1[0] = a; r2[0] = c; }
    }
    __syncthreads();
    float mean = r1[0] * (1.f / Hn);
    float var  = r2[0] * (1.f / Hn) - mean * mean;
    float inv  = rsqrtf(var + 1e-12f);
#pragma unroll
    for (int i = 0; i < 3; ++i) {
        int j = tid + i * 256;
        float v = (y[i] - mean) * inv * g[j] + bta[j];
        h [base + j] = v;
        hb[base + j] = f2bf(v);
    }
}

// ---------------------------------------------------------------------------
// WMMA GEMM: out[M,N] = A_bf16[M,K] @ WT_bf16[n][k] + bias (+ exact GELU)
// Block tile 128x128, 8 waves (2x4), wave tile 64x32 (4x2 wmma tiles).
// Double-buffered LDS staged with async global->LDS b128 copies.
// transV: bf16 output stored as vT[b, head, d, s] for attention PV.
// ---------------------------------------------------------------------------
__global__ __launch_bounds__(256)
void wmma_gemm_kernel(const unsigned short* __restrict__ A,
                      const unsigned short* __restrict__ WT,
                      const float* __restrict__ bias,
                      float* __restrict__ outF,
                      unsigned short* __restrict__ outB,
                      int N, int K, int gelu, int transV) {
    __shared__ unsigned short sA[2][128 * 32];   // [m][k]
    __shared__ unsigned short sB[2][128 * 32];   // [n][k]
    int tid = threadIdx.x, lane = tid & 31, wave = tid >> 5;
    int wm = wave >> 2, wn = wave & 3;
    int bm = blockIdx.y * 128, bn = blockIdx.x * 128;

    v8f acc[4][2];
#pragma unroll
    for (int m = 0; m < 4; ++m)
#pragma unroll
        for (int n = 0; n < 2; ++n) acc[m][n] = v8f_zero();

    // staging geometry: 512 b128 segments per 128x32 tile, 2 per thread
    int r0 = tid >> 2,           c0 = (tid & 3) * 8;
    int r1 = (tid + 256) >> 2,   c1 = (tid & 3) * 8;   // (tid+256)&3 == tid&3

    auto stage = [&](int buf, int k0) {
        async_b128(A  + (size_t)(bm + r0) * K + k0 + c0, &sA[buf][r0 * 32 + c0]);
        async_b128(A  + (size_t)(bm + r1) * K + k0 + c1, &sA[buf][r1 * 32 + c1]);
        async_b128(WT + (size_t)(bn + r0) * K + k0 + c0, &sB[buf][r0 * 32 + c0]);
        async_b128(WT + (size_t)(bn + r1) * K + k0 + c1, &sB[buf][r1 * 32 + c1]);
    };

    int KT = K >> 5;
    stage(0, 0);
    for (int kt = 0; kt < KT; ++kt) {
        int nxt = kt + 1;
        if (nxt < KT) { stage(nxt & 1, nxt * 32); wait_async4(); }
        else          { wait_async0(); }
        __syncthreads();
        const unsigned short* a_ = sA[kt & 1];
        const unsigned short* b_ = sB[kt & 1];
        v16bf bfr[2];
#pragma unroll
        for (int n = 0; n < 2; ++n)
            bfr[n] = load_frag(b_, 32, wn * 32 + n * 16, 0, lane);
#pragma unroll
        for (int m = 0; m < 4; ++m) {
            v16bf afr = load_frag(a_, 32, wm * 64 + m * 16, 0, lane);
#pragma unroll
            for (int n = 0; n < 2; ++n)
                acc[m][n] = wmma_bf16(afr, bfr[n], acc[m][n]);
        }
        __syncthreads();
    }

    // epilogue: C/D layout -> lane 0-15: N=lane, M=r; lane 16-31: M=r+8
    int rofs = (lane >> 4) * 8, cl = lane & 15;
#pragma unroll
    for (int m = 0; m < 4; ++m)
#pragma unroll
        for (int n = 0; n < 2; ++n) {
            int col = bn + wn * 32 + n * 16 + cl;
            if (col < N) {
                float bv = bias[col];
#pragma unroll
                for (int r = 0; r < 8; ++r) {
                    int row = bm + wm * 64 + m * 16 + rofs + r;
                    float val = acc[m][n][r] + bv;
                    if (gelu) val = 0.5f * val * (1.f + erff(val * 0.70710678118f));
                    if (outF) outF[(size_t)row * N + col] = val;
                    if (outB) {
                        size_t o;
                        if (transV)
                            o = (((size_t)(row >> 9) * NHn + (col >> 6)) * DHn
                                 + (col & 63)) * Sn + (row & 511);
                        else
                            o = (size_t)row * N + col;
                        outB[o] = f2bf(val);
                    }
                }
            }
        }
}

// ---------------------------------------------------------------------------
// Attention: one block (128 thr = 4 waves) per (b, head, 16-query tile).
// All LDS staging via async global->LDS b128 copies (Q, K chunks, vT chunks).
// ---------------------------------------------------------------------------
__global__ __launch_bounds__(128)
void attention_kernel(const unsigned short* __restrict__ Q,
                      const unsigned short* __restrict__ Km,
                      const unsigned short* __restrict__ vT,
                      const int* __restrict__ amask,
                      unsigned short* __restrict__ ctx) {
    __shared__ unsigned short sQ[16 * 64];       //  2 KB  [q][d]
    __shared__ unsigned short sKV[64 * 64];      //  8 KB  scores:[key][d] / ctx:[d][key]
    __shared__ float          sS[16 * 512];      // 32 KB  scores
    __shared__ unsigned short sP[16 * 512];      // 16 KB  probs bf16

    int tid = threadIdx.x, lane = tid & 31, wave = tid >> 5;
    int q0 = blockIdx.x * 16, head = blockIdx.y, b = blockIdx.z;
    const size_t base  = ((size_t)b * Sn) * Hn + head * DHn;
    const size_t vbase = ((size_t)(b * NHn + head)) * DHn * Sn;   // [d][s]

    { // Q tile: one async b128 per thread
        int r = tid >> 3, d = (tid & 7) * 8;
        async_b128(Q + base + (size_t)(q0 + r) * Hn + d, &sQ[r * 64 + d]);
    }

    // ---- scores: 8 key chunks of 64 ----
    for (int c = 0; c < 8; ++c) {
        __syncthreads();
#pragma unroll
        for (int i = 0; i < 4; ++i) {
            int seg = i * 128 + tid;             // 0..511
            int r = seg >> 3, d = (seg & 7) * 8;
            async_b128(Km + base + (size_t)(c * 64 + r) * Hn + d, &sKV[r * 64 + d]);
        }
        wait_async0();
        __syncthreads();
        v8f acc = v8f_zero();
#pragma unroll
        for (int ks = 0; ks < 2; ++ks) {
            v16bf a  = load_frag(sQ, 64, 0, ks * 32, lane);
            v16bf bb = load_frag(sKV, 64, wave * 16, ks * 32, lane);
            acc = wmma_bf16(a, bb, acc);
        }
        int key = c * 64 + wave * 16 + (lane & 15);
        float mb = amask[b * Sn + key] ? 0.f : -1e9f;
        int rofs = (lane >> 4) * 8;
#pragma unroll
        for (int r = 0; r < 8; ++r)
            sS[(rofs + r) * 512 + key] = acc[r] * 0.125f + mb;
    }
    __syncthreads();

    // ---- softmax: 16 rows x 8 threads/row ----
    {
        int row = tid >> 3, sub = tid & 7;
        float mx = -1e30f;
        for (int j = sub; j < 512; j += 8) mx = fmaxf(mx, sS[row * 512 + j]);
#pragma unroll
        for (int o = 4; o; o >>= 1) mx = fmaxf(mx, __shfl_xor(mx, o, 8));
        float sum = 0.f;
        for (int j = sub; j < 512; j += 8) {
            float e = __expf(sS[row * 512 + j] - mx);
            sS[row * 512 + j] = e; sum += e;
        }
#pragma unroll
        for (int o = 4; o; o >>= 1) sum += __shfl_xor(sum, o, 8);
        float inv = 1.f / sum;
        for (int j = sub; j < 512; j += 8)
            sP[row * 512 + j] = f2bf(sS[row * 512 + j] * inv);
    }

    // ---- ctx = P @ V : accumulate over 8 chunks (vT is [d][s], contiguous) ----
    v8f acc = v8f_zero();
    for (int c = 0; c < 8; ++c) {
        __syncthreads();
#pragma unroll
        for (int i = 0; i < 4; ++i) {
            int seg = i * 128 + tid;             // 0..511
            int d = seg >> 3, k0 = (seg & 7) * 8;
            async_b128(vT + vbase + (size_t)d * Sn + c * 64 + k0, &sKV[d * 64 + k0]);
        }
        wait_async0();
        __syncthreads();
#pragma unroll
        for (int ks = 0; ks < 2; ++ks) {
            v16bf a  = load_frag(sP, 512, 0, c * 64 + ks * 32, lane);
            v16bf bb = load_frag(sKV, 64, wave * 16, ks * 32, lane);
            acc = wmma_bf16(a, bb, acc);
        }
    }
    int d = wave * 16 + (lane & 15);
    int rofs = (lane >> 4) * 8;
#pragma unroll
    for (int r = 0; r < 8; ++r)
        ctx[base + (size_t)(q0 + rofs + r) * Hn + d] = f2bf(acc[r]);
}

// ---------------------------------------------------------------------------
// Classifier mask + softmax over 61 codons (1 wave / token)
// ---------------------------------------------------------------------------
__global__ __launch_bounds__(32)
void mask_softmax_kernel(const float* __restrict__ logits,   // [M][61]
                         const int* __restrict__ x,
                         const int* __restrict__ seq_lens,
                         const int* __restrict__ mflag,
                         const float* __restrict__ table,    // [25][61]
                         float* __restrict__ out) {
    int tok = blockIdx.x, b = tok >> 9, s = tok & 511, tid = threadIdx.x;
    int doMask = *mflag;
    int id = x[tok];
    bool valid = s < seq_lens[b];
    int c0 = tid, c1 = tid + 32;
    float v0 = -1e30f, v1 = -1e30f;
    if (c0 < NC) { v0 = logits[(size_t)tok * NC + c0];
                   if (doMask) v0 += valid ? table[id * NC + c0] : -1e9f; }
    if (c1 < NC) { v1 = logits[(size_t)tok * NC + c1];
                   if (doMask) v1 += valid ? table[id * NC + c1] : -1e9f; }
    float mx = fmaxf(v0, v1);
#pragma unroll
    for (int o = 16; o; o >>= 1) mx = fmaxf(mx, __shfl_xor(mx, o, 32));
    float e0 = (c0 < NC) ? __expf(v0 - mx) : 0.f;
    float e1 = (c1 < NC) ? __expf(v1 - mx) : 0.f;
    float sum = e0 + e1;
#pragma unroll
    for (int o = 16; o; o >>= 1) sum += __shfl_xor(sum, o, 32);
    float inv = 1.f / sum;
    if (c0 < NC) out[(size_t)tok * NC + c0] = e0 * inv;
    if (c1 < NC) out[(size_t)tok * NC + c1] = e1 * inv;
}

// ---------------------------------------------------------------------------
// Host launcher
// ---------------------------------------------------------------------------
extern "C" void kernel_launch(void* const* d_in, const int* in_sizes, int n_in,
                              void* d_out, int out_size, void* d_ws, size_t ws_size,
                              hipStream_t stream) {
    (void)in_sizes; (void)n_in; (void)out_size; (void)ws_size;
    const int*   x          = (const int*)  d_in[0];
    const int*   seq_lens   = (const int*)  d_in[1];
    const int*   attn_mask  = (const int*)  d_in[2];
    const int*   mflag      = (const int*)  d_in[3];
    const float* word_emb   = (const float*)d_in[4];
    const float* pos_emb    = (const float*)d_in[5];
    const float* emb_g      = (const float*)d_in[6];
    const float* emb_b      = (const float*)d_in[7];
    const float* wq         = (const float*)d_in[8];
    const float* bq         = (const float*)d_in[9];
    const float* wk         = (const float*)d_in[10];
    const float* bk         = (const float*)d_in[11];
    const float* wv         = (const float*)d_in[12];
    const float* bv         = (const float*)d_in[13];
    const float* wo         = (const float*)d_in[14];
    const float* bo         = (const float*)d_in[15];
    const float* ln1g       = (const float*)d_in[16];
    const float* ln1b       = (const float*)d_in[17];
    const float* w1         = (const float*)d_in[18];
    const float* b1         = (const float*)d_in[19];
    const float* w2         = (const float*)d_in[20];
    const float* b2         = (const float*)d_in[21];
    const float* ln2g       = (const float*)d_in[22];
    const float* ln2b       = (const float*)d_in[23];
    const float* w_cls      = (const float*)d_in[24];
    const float* b_cls      = (const float*)d_in[25];
    const float* mask_table = (const float*)d_in[26];

    // workspace layout
    size_t off = 0;
    auto alloc = [&](size_t bytes) {
        void* p = (char*)d_ws + off;
        off += (bytes + 255) & ~(size_t)255;
        return p;
    };
    float*          h      = (float*)alloc((size_t)Mn * Hn * 4);
    unsigned short* hb     = (unsigned short*)alloc((size_t)Mn * Hn * 2);
    unsigned short* qb     = (unsigned short*)alloc((size_t)Mn * Hn * 2);
    unsigned short* kb_    = (unsigned short*)alloc((size_t)Mn * Hn * 2);
    unsigned short* vtb    = (unsigned short*)alloc((size_t)Mn * Hn * 2);  // [b,h,d,s]
    unsigned short* ctxb   = (unsigned short*)alloc((size_t)Mn * Hn * 2);
    float*          tmp    = (float*)alloc((size_t)Mn * Hn * 4);
    unsigned short* ffb    = (unsigned short*)alloc((size_t)Mn * FFn * 2);
    float*          logits = (float*)alloc((size_t)Mn * NC * 4);
    unsigned short* wtb    = (unsigned short*)alloc((size_t)FFn * Hn * 2); // WT scratch

    auto gemm = [&](const unsigned short* Abuf, const float* W, const float* bias,
                    float* oF, unsigned short* oB, int N, int K, int act, int trv) {
        int Npad = (N + 127) & ~127;
        convert_wt_kernel<<<(unsigned)(((size_t)Npad * K) / 256), 256, 0, stream>>>(
            W, wtb, N, K);
        dim3 grid(Npad / 128, Mn / 128);
        wmma_gemm_kernel<<<grid, 256, 0, stream>>>(Abuf, wtb, bias, oF, oB,
                                                   N, K, act, trv);
    };

    embed_ln_kernel<<<Mn, 256, 0, stream>>>(x, word_emb, pos_emb, emb_g, emb_b, h, hb);

    for (int l = 0; l < Ln; ++l) {
        const float* wq_l = wq + (size_t)l * Hn * Hn;
        const float* wk_l = wk + (size_t)l * Hn * Hn;
        const float* wv_l = wv + (size_t)l * Hn * Hn;
        const float* wo_l = wo + (size_t)l * Hn * Hn;
        const float* w1_l = w1 + (size_t)l * Hn * FFn;
        const float* w2_l = w2 + (size_t)l * FFn * Hn;

        gemm(hb, wq_l, bq + l * Hn, nullptr, qb,  Hn, Hn, 0, 0);
        gemm(hb, wk_l, bk + l * Hn, nullptr, kb_, Hn, Hn, 0, 0);
        gemm(hb, wv_l, bv + l * Hn, nullptr, vtb, Hn, Hn, 0, 1);   // -> vT layout

        attention_kernel<<<dim3(Sn / 16, NHn, Bn), 128, 0, stream>>>(
            qb, kb_, vtb, attn_mask, ctxb);

        gemm(ctxb, wo_l, bo + l * Hn, tmp, nullptr, Hn, Hn, 0, 0);
        add_ln_kernel<<<Mn, 256, 0, stream>>>(h, tmp, ln1g + l * Hn, ln1b + l * Hn, h, hb);

        gemm(hb, w1_l, b1 + l * FFn, nullptr, ffb, FFn, Hn, 1, 0); // + GELU
        gemm(ffb, w2_l, b2 + l * Hn, tmp, nullptr, Hn, FFn, 0, 0);
        add_ln_kernel<<<Mn, 256, 0, stream>>>(h, tmp, ln2g + l * Hn, ln2b + l * Hn, h, hb);
    }

    gemm(hb, w_cls, b_cls, logits, nullptr, NC, Hn, 0, 0);

    mask_softmax_kernel<<<Mn, 32, 0, stream>>>(logits, x, seq_lens, mflag,
                                               mask_table, (float*)d_out);
}
